// MultiHeadAttention_30502857736900
// MI455X (gfx1250) — matrix operand order repaired and checked by reading, not compiled, not run on hardware
//
#include <hip/hip_runtime.h>
#include <hip/hip_bf16.h>

// ---------------------------------------------------------------------------
// MHA forward for MI455X (gfx1250, wave32, WMMA).
// Pipeline: fp32->bf16 convert -> QKV WMMA GEMMs -> flash attention (async
// global->LDS staged K/V tiles, WMMA QK^T + online softmax + WMMA PV) ->
// output-projection WMMA GEMM + bias.
// All matmul FLOPs go through v_wmma_f32_16x16x32_bf16 (fp32 accumulate).
// K/V tiles in attention are staged with GLOBAL_LOAD_ASYNC_TO_LDS_B128
// (ASYNCcnt) so the 4 waves of a block share one copy instead of 4x loads.
// ---------------------------------------------------------------------------

typedef __bf16 bf16_t;
typedef __attribute__((ext_vector_type(16))) __bf16 v16bf;
typedef __attribute__((ext_vector_type(8)))  __bf16 v8bf;
typedef __attribute__((ext_vector_type(4)))  __bf16 v4bf;
typedef __attribute__((ext_vector_type(8)))  float  v8f;

#define BATCH    4
#define SEQLEN   2048
#define DMODEL   1024
#define NHEADS   16
#define HDIM     64

// ---------------------------------------------------------------------------
// Fragment loaders (layouts per CDNA5 ISA 7.12.2)
//
// A (16x32 bf16, MxK), source row-major [M][K]:
//   lanes 0-15 : row M=lane,  elems 0..7 = K k0+0..7,  elems 8..15 = K k0+16..23
//   lanes 16-31: row M=ln-16, elems 0..7 = K k0+8..15, elems 8..15 = K k0+24..31
__device__ __forceinline__
v16bf load_a_frag(const bf16_t* base, int ld, int row0, int k0, int lane) {
  const int half = lane >> 4;
  const int r    = row0 + (lane & 15);
  const bf16_t* p = base + (size_t)r * ld + k0 + half * 8;
  v8bf lo = *(const v8bf*)(p);
  v8bf hi = *(const v8bf*)(p + 16);
  return __builtin_shufflevector(lo, hi, 0,1,2,3,4,5,6,7,8,9,10,11,12,13,14,15);
}

// B (32x16 bf16, KxN), source stored as [N][K] row-major (B^T storage):
//   lanes 0-15 : col N=lane,  elems 0..7 = K k0+0..7,  elems 8..15 = K k0+16..23
//   lanes 16-31: col N=ln-16, elems 0..7 = K k0+8..15, elems 8..15 = K k0+24..31
// => two 16-byte loads per lane.
__device__ __forceinline__
v16bf load_b_frag(const bf16_t* base, int ld, int n0, int k0, int lane) {
  const int khalf = lane >> 4;
  const int n     = n0 + (lane & 15);
  const bf16_t* p = base + (size_t)n * ld + k0 + khalf * 8;
  v8bf lo = *(const v8bf*)(p);
  v8bf hi = *(const v8bf*)(p + 16);
  return __builtin_shufflevector(lo, hi, 0,1,2,3,4,5,6,7,8,9,10,11,12,13,14,15);
}

__device__ __forceinline__
v8f wmma_bf16(v16bf a, v16bf b, v8f c) {
  return __builtin_amdgcn_wmma_f32_16x16x32_bf16(false, a, false, b, (short)0, c, false, false);
}

// ---------------------------------------------------------------------------
// CDNA5 async global -> LDS copy, 16 bytes per lane, tracked by ASYNCcnt.
// (CDNA5 ISA: GLOBAL_LOAD_ASYNC_TO_LDS_B128, GV mode)
__device__ __forceinline__
void async_g2lds_b128(unsigned lds_addr, const void* gptr) {
  asm volatile("global_load_async_to_lds_b128 %0, %1, off"
               :: "v"(lds_addr), "v"(gptr) : "memory");
}
__device__ __forceinline__ void wait_asynccnt0() {
  asm volatile("s_wait_asynccnt 0x0" ::: "memory");
}

// ---------------------------------------------------------------------------
// fp32 -> bf16 conversion (n multiple of 4)
__global__ __launch_bounds__(256)
void cvt_f32_bf16(const float* __restrict__ src, bf16_t* __restrict__ dst, int n) {
  int i = (blockIdx.x * 256 + threadIdx.x) * 4;
  if (i < n) {
    float4 f = *(const float4*)(src + i);
    v4bf o;
    o.x = (bf16_t)f.x; o.y = (bf16_t)f.y; o.z = (bf16_t)f.z; o.w = (bf16_t)f.w;
    *(v4bf*)(dst + i) = o;
  }
}

// ---------------------------------------------------------------------------
// QKV projection GEMM: P[m][n] = sum_k X[m][k] * W[n][k]   (M=8192, N=K=1024)
// Block = 128 threads (4 waves), block tile 128x128, wave tile 64x64.
__global__ __launch_bounds__(128)
void gemm_qkv_kernel(const bf16_t* __restrict__ X,   // [8192][1024]
                     const bf16_t* __restrict__ W,   // [1024][1024]
                     bf16_t* __restrict__ dst,
                     int transposed) {               // 0: [b,h,t,d]  1: [b,h,d,t]
  const int lane = threadIdx.x & 31;
  const int wave = threadIdx.x >> 5;
  const int m0 = blockIdx.x * 128 + (wave & 1) * 64;
  const int n0 = blockIdx.y * 128 + (wave >> 1) * 64;

  v8f acc[4][4] = {};
  for (int k0 = 0; k0 < DMODEL; k0 += 32) {
    v16bf a[4], b[4];
#pragma unroll
    for (int i = 0; i < 4; ++i) a[i] = load_a_frag(X, DMODEL, m0 + i * 16, k0, lane);
#pragma unroll
    for (int j = 0; j < 4; ++j) b[j] = load_b_frag(W, DMODEL, n0 + j * 16, k0, lane);
#pragma unroll
    for (int i = 0; i < 4; ++i)
#pragma unroll
      for (int j = 0; j < 4; ++j)
        acc[i][j] = wmma_bf16(a[i], b[j], acc[i][j]);
  }

  // D layout: elem e of accum -> row = 8*(lane>>4)+e, col = lane&15
  const int col = lane & 15;
  const int rh  = lane >> 4;
#pragma unroll
  for (int i = 0; i < 4; ++i)
#pragma unroll
    for (int j = 0; j < 4; ++j)
#pragma unroll
      for (int e = 0; e < 8; ++e) {
        const int m = m0 + i * 16 + rh * 8 + e;
        const int n = n0 + j * 16 + col;
        const int bb = m >> 11, t = m & (SEQLEN - 1);
        const int h = n >> 6, d = n & (HDIM - 1);
        const bf16_t v = (bf16_t)acc[i][j][e];
        if (!transposed)
          dst[(((size_t)(bb * NHEADS + h)) * SEQLEN + t) * HDIM + d] = v;
        else
          dst[(((size_t)(bb * NHEADS + h)) * HDIM + d) * SEQLEN + t] = v;
      }
}

// ---------------------------------------------------------------------------
// Flash attention, causal. Grid (SEQ/64, B*H), block = 128 (4 waves).
// Each wave owns 16 query rows; workgroup covers 64 queries.
// K/V tiles for each 64-key block are staged once per block into LDS with
// async global->LDS loads; all 4 waves read fragments from LDS.
__global__ __launch_bounds__(128)
void attn_kernel(const bf16_t* __restrict__ Qb,   // [BH][T][64]
                 const bf16_t* __restrict__ Kb,   // [BH][T][64]
                 const bf16_t* __restrict__ Vt,   // [BH][64][T]
                 bf16_t* __restrict__ Ctx) {      // [B][T][1024]
  __shared__ __align__(32) bf16_t Klds[64 * HDIM];    // [key][hd]     8KB
  __shared__ __align__(32) bf16_t Vlds[HDIM * 64];    // [hd][key]     8KB
  __shared__ __align__(32) bf16_t Pld[4][16][HDIM];   // per-wave P    8KB

  const int tid  = threadIdx.x;
  const int lane = tid & 31;
  const int wave = tid >> 5;
  const int bh   = blockIdx.y;
  const int qblk = blockIdx.x;
  const int q0   = qblk * 64 + wave * 16;

  const bf16_t* Q = Qb + (size_t)bh * SEQLEN * HDIM;
  const bf16_t* K = Kb + (size_t)bh * SEQLEN * HDIM;
  const bf16_t* V = Vt + (size_t)bh * HDIM * SEQLEN;

  const unsigned kldsBase = (unsigned)(unsigned long long)(&Klds[0]);
  const unsigned vldsBase = (unsigned)(unsigned long long)(&Vlds[0]);

  // Q fragments are reused across all key blocks
  const v16bf qa0 = load_a_frag(Q, HDIM, q0, 0, lane);
  const v16bf qa1 = load_a_frag(Q, HDIM, q0, 32, lane);

  v8f   o[4] = {};
  float mrow[8], lrow[8];
#pragma unroll
  for (int e = 0; e < 8; ++e) { mrow[e] = -3.0e38f; lrow[e] = 0.0f; }

  const int   col   = lane & 15;
  const int   rh    = lane >> 4;
  const float scale = 0.125f;   // 1/sqrt(HDIM)

  for (int kb = 0; kb <= qblk; ++kb) {
    const int kbase = kb * 64;

    // ---- async-stage K tile (contiguous 8KB) and V tile (64 x 128B rows) ----
    // 512 16-byte chunks per tile, 128 threads -> 4 chunks/thread/tile.
#pragma unroll
    for (int i = 0; i < 4; ++i) {
      const int c = tid + i * 128;                  // chunk 0..511
      async_g2lds_b128(kldsBase + c * 16,
                       (const char*)K + (size_t)kbase * (HDIM * 2) + c * 16);
      const int r = c >> 2, s = c & 3;              // V: row=hd, 4 chunks/row
      async_g2lds_b128(vldsBase + (unsigned)(r * HDIM + s * 16) * 2,
                       (const char*)V + ((size_t)r * SEQLEN + kbase + s * 16) * 2);
    }
    wait_asynccnt0();
    __syncthreads();                                // tiles visible to all waves

    // ---- S = Q @ K^T (16 x 64), HDIM contraction in 2 steps of 32 ----
    v8f s[4] = {};
#pragma unroll
    for (int j = 0; j < 4; ++j) {
      v16bf b0 = load_b_frag(Klds, HDIM, j * 16, 0, lane);
      v16bf b1 = load_b_frag(Klds, HDIM, j * 16, 32, lane);
      s[j] = wmma_bf16(qa0, b0, s[j]);
      s[j] = wmma_bf16(qa1, b1, s[j]);
    }

    // ---- scale + causal mask ----
#pragma unroll
    for (int j = 0; j < 4; ++j)
#pragma unroll
      for (int e = 0; e < 8; ++e) {
        const int qrow = q0 + rh * 8 + e;
        const int kcol = kbase + j * 16 + col;
        const float v = s[j][e] * scale;
        s[j][e] = (kcol <= qrow) ? v : -3.0e38f;
      }

    // ---- online softmax: row max across 4 tiles + 16 lanes of this half ----
    float rescale[8];
#pragma unroll
    for (int e = 0; e < 8; ++e) {
      float v = fmaxf(fmaxf(s[0][e], s[1][e]), fmaxf(s[2][e], s[3][e]));
#pragma unroll
      for (int msk = 1; msk < 16; msk <<= 1) v = fmaxf(v, __shfl_xor(v, msk, 32));
      const float mnew = fmaxf(mrow[e], v);
      rescale[e] = __expf(mrow[e] - mnew);
      mrow[e] = mnew;
    }

    // ---- P = exp(S - m); row sums; rescale O and l ----
#pragma unroll
    for (int e = 0; e < 8; ++e) {
      float sum = 0.f;
#pragma unroll
      for (int j = 0; j < 4; ++j) {
        const float p = __expf(s[j][e] - mrow[e]);
        s[j][e] = p;
        sum += p;
      }
#pragma unroll
      for (int msk = 1; msk < 16; msk <<= 1) sum += __shfl_xor(sum, msk, 32);
      lrow[e] = lrow[e] * rescale[e] + sum;
#pragma unroll
      for (int j = 0; j < 4; ++j) o[j][e] *= rescale[e];
    }

    // ---- stage P (D-layout regs) -> LDS row-major, reload as A fragments ----
#pragma unroll
    for (int j = 0; j < 4; ++j)
#pragma unroll
      for (int e = 0; e < 8; ++e)
        Pld[wave][rh * 8 + e][j * 16 + col] = (bf16_t)s[j][e];
    __syncthreads();   // uniform: all waves run the same kb trip count

    const v16bf pa0 = load_a_frag(&Pld[wave][0][0], HDIM, 0, 0, lane);
    const v16bf pa1 = load_a_frag(&Pld[wave][0][0], HDIM, 0, 32, lane);

    // ---- O += P @ V : B operand from LDS-staged transposed V tile ----
#pragma unroll
    for (int j = 0; j < 4; ++j) {
      v16bf bv0 = load_b_frag(Vlds, 64, j * 16, 0, lane);
      v16bf bv1 = load_b_frag(Vlds, 64, j * 16, 32, lane);
      o[j] = wmma_bf16(pa0, bv0, o[j]);
      o[j] = wmma_bf16(pa1, bv1, o[j]);
    }
    __syncthreads();   // protect K/V/P tiles before next iteration's staging
  }

  // ---- finalize: ctx = O / l, merge heads -> [b][t][h*64 + d] (bf16) ----
  const int bb = bh >> 4, h = bh & 15;
#pragma unroll
  for (int j = 0; j < 4; ++j)
#pragma unroll
    for (int e = 0; e < 8; ++e) {
      const int qrow = q0 + rh * 8 + e;
      const float val = o[j][e] / lrow[e];
      Ctx[((size_t)bb * SEQLEN + qrow) * DMODEL + h * HDIM + j * 16 + col] = (bf16_t)val;
    }
}

// ---------------------------------------------------------------------------
// Output projection: out[m][n] = sum_k Ctx[m][k]*Wo[n][k] + bias[n]  (fp32 out)
__global__ __launch_bounds__(128)
void gemm_out_kernel(const bf16_t* __restrict__ Ctx,  // [8192][1024]
                     const bf16_t* __restrict__ W,    // [1024][1024]
                     const float* __restrict__ bias,  // [1024]
                     float* __restrict__ out) {       // [8192][1024]
  const int lane = threadIdx.x & 31;
  const int wave = threadIdx.x >> 5;
  const int m0 = blockIdx.x * 128 + (wave & 1) * 64;
  const int n0 = blockIdx.y * 128 + (wave >> 1) * 64;

  v8f acc[4][4] = {};
  for (int k0 = 0; k0 < DMODEL; k0 += 32) {
    v16bf a[4], b[4];
#pragma unroll
    for (int i = 0; i < 4; ++i) a[i] = load_a_frag(Ctx, DMODEL, m0 + i * 16, k0, lane);
#pragma unroll
    for (int j = 0; j < 4; ++j) b[j] = load_b_frag(W, DMODEL, n0 + j * 16, k0, lane);
#pragma unroll
    for (int i = 0; i < 4; ++i)
#pragma unroll
      for (int j = 0; j < 4; ++j)
        acc[i][j] = wmma_bf16(a[i], b[j], acc[i][j]);
  }

  const int col = lane & 15;
  const int rh  = lane >> 4;
#pragma unroll
  for (int i = 0; i < 4; ++i)
#pragma unroll
    for (int j = 0; j < 4; ++j) {
      const int n = n0 + j * 16 + col;
      const float bv = bias[n];
#pragma unroll
      for (int e = 0; e < 8; ++e) {
        const int m = m0 + i * 16 + rh * 8 + e;
        out[(size_t)m * DMODEL + n] = acc[i][j][e] + bv;
      }
    }
}

// ---------------------------------------------------------------------------
extern "C" void kernel_launch(void* const* d_in, const int* in_sizes, int n_in,
                              void* d_out, int out_size, void* d_ws, size_t ws_size,
                              hipStream_t stream) {
  const float* X  = (const float*)d_in[0];
  const float* Wq = (const float*)d_in[1];
  const float* Wk = (const float*)d_in[2];
  const float* Wv = (const float*)d_in[3];
  const float* Wo = (const float*)d_in[4];
  const float* bo = (const float*)d_in[5];
  float* out = (float*)d_out;

  const size_t NX = (size_t)BATCH * SEQLEN * DMODEL;  // 8,388,608
  const size_t NW = (size_t)DMODEL * DMODEL;          // 1,048,576

  bf16_t* Xb   = (bf16_t*)d_ws;
  bf16_t* Wqb  = Xb   + NX;
  bf16_t* Wkb  = Wqb  + NW;
  bf16_t* Wvb  = Wkb  + NW;
  bf16_t* Wob  = Wvb  + NW;
  bf16_t* Qb   = Wob  + NW;   // [B,H,T,64]
  bf16_t* Kb   = Qb   + NX;   // [B,H,T,64]
  bf16_t* Vtb  = Kb   + NX;   // [B,H,64,T]
  bf16_t* Ctxb = Vtb  + NX;   // [B,T,1024]
  (void)ws_size; (void)in_sizes; (void)n_in; (void)out_size;

  // 1) fp32 -> bf16
  cvt_f32_bf16<<<dim3((unsigned)(NX / 1024)), 256, 0, stream>>>(X,  Xb,  (int)NX);
  cvt_f32_bf16<<<dim3((unsigned)(NW / 1024)), 256, 0, stream>>>(Wq, Wqb, (int)NW);
  cvt_f32_bf16<<<dim3((unsigned)(NW / 1024)), 256, 0, stream>>>(Wk, Wkb, (int)NW);
  cvt_f32_bf16<<<dim3((unsigned)(NW / 1024)), 256, 0, stream>>>(Wv, Wvb, (int)NW);
  cvt_f32_bf16<<<dim3((unsigned)(NW / 1024)), 256, 0, stream>>>(Wo, Wob, (int)NW);

  // 2) QKV projections (M=8192, N=1024 -> grid 64 x 8, 128 threads)
  dim3 ggrid(64, 8);
  gemm_qkv_kernel<<<ggrid, 128, 0, stream>>>(Xb, Wqb, Qb, 0);
  gemm_qkv_kernel<<<ggrid, 128, 0, stream>>>(Xb, Wkb, Kb, 0);
  gemm_qkv_kernel<<<ggrid, 128, 0, stream>>>(Xb, Wvb, Vtb, 1);

  // 3) causal flash attention: grid (T/64, B*H)
  attn_kernel<<<dim3(SEQLEN / 64, BATCH * NHEADS), 128, 0, stream>>>(Qb, Kb, Vtb, Ctxb);

  // 4) output projection + bias
  gemm_out_kernel<<<ggrid, 128, 0, stream>>>(Ctxb, Wob, bo, out);
}
